// LSA_73710228734879
// MI455X (gfx1250) — matrix operand order repaired and checked by reading, not compiled, not run on hardware
//
#include <hip/hip_runtime.h>
#include <hip/hip_bf16.h>

#define B_      8
#define N_      1025
#define NPAD    1056          // 66 tiles of 16, multiple of 32 for j-steps
#define DIM_    512
#define HEADS_  12
#define DHEAD   64
#define INNER_  768
#define QKV3    2304
#define M_TOT   (B_ * N_)     // 8200
#define HW_     32

typedef __attribute__((ext_vector_type(16))) _Float16 v16h;
typedef __attribute__((ext_vector_type(8)))  _Float16 v8h;
typedef __attribute__((ext_vector_type(8)))  float    v8f;

#if defined(__has_builtin)
#if __has_builtin(__builtin_amdgcn_global_load_async_to_lds_b128)
#define HAVE_ASYNC_LDS 1
#endif
#if __has_builtin(__builtin_amdgcn_s_wait_asynccnt)
#define HAVE_WAIT_ASYNC 1
#endif
#endif

// 16-byte global -> LDS copy; async hardware path on gfx1250 when available.
// Inline asm avoids the builtin's unspellable __device__/__shared__ LangAS
// pointer parameters. VDST = LDS byte address VGPR, VADDR = 64-bit global.
__device__ __forceinline__ void async_cp16(_Float16* dst, const _Float16* src) {
#if HAVE_ASYNC_LDS
  unsigned ldsoff = (unsigned)(uintptr_t)dst;
  asm volatile("global_load_async_to_lds_b128 %0, %1, off"
               :: "v"(ldsoff), "v"(src)
               : "memory");
#else
  *(v8h*)dst = *(const v8h*)src;
#endif
}

__device__ __forceinline__ void wait_async_lds() {
#if HAVE_ASYNC_LDS
#if HAVE_WAIT_ASYNC
  __builtin_amdgcn_s_wait_asynccnt(0);
#else
  asm volatile("s_wait_asynccnt 0x0" ::: "memory");
#endif
#endif
}

__device__ __forceinline__ v8f wmma16(v16h a, v16h b, v8f c) {
  // D = A(16x32 f16) * B(32x16 f16) + C(16x16 f32)
  return __builtin_amdgcn_wmma_f32_16x16x32_f16(
      false, a, false, b, (short)0, c, false, false);
}

// A-operand fragment from an f32 row-major matrix (convert to f16).
// 16-bit A 16x32 layout: lane = l, row M = l&15, half hi = l>>4.
// K(e) = hi*8 + e (e<8); K(e) = 16 + hi*8 + (e-8) (e>=8) -> two 8-float runs.
__device__ __forceinline__ v16h frag_a_f32(const float* __restrict__ mat, int ld,
                                           int m0, int mmax, int k0) {
  int l = threadIdx.x & 31;
  int r = l & 15, hi = l >> 4;
  int row = m0 + r; if (row >= mmax) row = mmax - 1;
  const float* p = mat + (size_t)row * ld + k0 + hi * 8;
  v16h a;
#pragma unroll
  for (int e = 0; e < 16; ++e) a[e] = (_Float16)p[(e < 8) ? e : (e + 8)];
  return a;
}

// A-operand fragment from an f16 row-major matrix (16B-aligned rows).
__device__ __forceinline__ v16h frag_a_f16(const _Float16* __restrict__ mat, int ld,
                                           int m0, int k0) {
  int l = threadIdx.x & 31;
  int r = l & 15, hi = l >> 4;
  const _Float16* p = mat + (size_t)(m0 + r) * ld + k0 + hi * 8;
  v8h x0 = *(const v8h*)p;
  v8h x1 = *(const v8h*)(p + 16);
  v16h a;
#pragma unroll
  for (int e = 0; e < 8; ++e) { a[e] = x0[e]; a[e + 8] = x1[e]; }
  return a;
}

// B-operand fragment: mat holds B^T row-major (row n = output column n).
// B 32x16 layout: lane column N = l&15; K(e) = hi*8 + e (e<8); K(e) = 16 + hi*8 + (e-8) (e>=8).
__device__ __forceinline__ v16h frag_b_f32(const float* __restrict__ mat, int ld,
                                           int n0, int k0) {
  int l = threadIdx.x & 31;
  int n = l & 15, hi = l >> 4;
  const float* p = mat + (size_t)(n0 + n) * ld + k0 + hi * 8;
  v16h b;
#pragma unroll
  for (int e = 0; e < 16; ++e) b[e] = (_Float16)p[(e < 8) ? e : (e + 8)];
  return b;
}

// Works for both global f16 and LDS f16 (inlined; compiler picks ds/global ops).
__device__ __forceinline__ v16h frag_b_f16(const _Float16* mat, int ld,
                                           int n0, int k0) {
  int l = threadIdx.x & 31;
  int n = l & 15, hi = l >> 4;
  const _Float16* p = mat + (size_t)(n0 + n) * ld + k0 + hi * 8;
  v8h x0 = *(const v8h*)p;
  v8h x1 = *(const v8h*)(p + 16);
  v16h b;
#pragma unroll
  for (int e = 0; e < 8; ++e) { b[e] = x0[e]; b[e + 8] = x1[e]; }
  return b;
}

// ---------------- Kernel 1: fused QKV projection (f32 -> f16 WMMA GEMM) ------
// qkv = x(8200x512) * w_qkv^T(512x2304); each wave owns a 32x64 output block.
__global__ __launch_bounds__(256) void qkv_kernel(
    const float* __restrict__ x, const float* __restrict__ w_qkv,
    _Float16* __restrict__ qh, _Float16* __restrict__ kh,
    _Float16* __restrict__ vT) {
  const int NG = QKV3 / 64;                   // 36 n-groups
  const int MG = (M_TOT + 31) / 32;           // 257 m-groups
  int wave = threadIdx.x >> 5;
  int id   = blockIdx.x * 8 + wave;
  if (id >= NG * MG) return;
  int ng = id % NG, mg = id / NG;
  int m0 = mg * 32, n0 = ng * 64;
  int l  = threadIdx.x & 31;
  v8f c[2][4] = {};
  for (int k0 = 0; k0 < DIM_; k0 += 32) {
    if (k0 + 64 < DIM_) {
      int prow = m0 + (l & 15); if (prow >= M_TOT) prow = M_TOT - 1;
      __builtin_prefetch(x + (size_t)prow * DIM_ + k0 + 64, 0, 1);
    }
    // preload all fragments so the 8 WMMAs issue without WAR hazard stalls
    v16h a0 = frag_a_f32(x, DIM_, m0, M_TOT, k0);
    v16h a1 = frag_a_f32(x, DIM_, m0 + 16, M_TOT, k0);
    v16h b0 = frag_b_f32(w_qkv, DIM_, n0,      k0);
    v16h b1 = frag_b_f32(w_qkv, DIM_, n0 + 16, k0);
    v16h b2 = frag_b_f32(w_qkv, DIM_, n0 + 32, k0);
    v16h b3 = frag_b_f32(w_qkv, DIM_, n0 + 48, k0);
    c[0][0] = wmma16(a0, b0, c[0][0]);
    c[1][0] = wmma16(a1, b0, c[1][0]);
    c[0][1] = wmma16(a0, b1, c[0][1]);
    c[1][1] = wmma16(a1, b1, c[1][1]);
    c[0][2] = wmma16(a0, b2, c[0][2]);
    c[1][2] = wmma16(a1, b2, c[1][2]);
    c[0][3] = wmma16(a0, b3, c[0][3]);
    c[1][3] = wmma16(a1, b3, c[1][3]);
  }
  int hi = l >> 4;
  const float scale = 0.125f;                 // DIM_HEAD^-0.5
#pragma unroll
  for (int i = 0; i < 2; ++i) {
#pragma unroll
    for (int j = 0; j < 4; ++j) {
      int col = n0 + j * 16 + (l & 15);
      int which = col / INNER_;               // 0=q 1=k 2=v
      int oc    = col - which * INNER_;
      int h = oc >> 6;
      int d = oc & 63;
#pragma unroll
      for (int r = 0; r < 8; ++r) {
        int m = m0 + i * 16 + r + hi * 8;
        if (m >= M_TOT) continue;
        int bidx = m / N_;
        int nn   = m - bidx * N_;
        size_t hb = (size_t)bidx * HEADS_ + h;
        float val = c[i][j][r];
        if (which == 0)      qh[(hb * NPAD + nn) * DHEAD + d] = (_Float16)(val * scale);
        else if (which == 1) kh[(hb * NPAD + nn) * DHEAD + d] = (_Float16)val;
        else                 vT[(hb * DHEAD + d) * NPAD + nn] = (_Float16)val;
      }
    }
  }
}

// ---------------- Kernel 2: flash attention, double-buffered async K/V staging
__global__ __launch_bounds__(256) void attn_kernel(
    const _Float16* __restrict__ qh, const _Float16* __restrict__ kh,
    const _Float16* __restrict__ vT, float* __restrict__ out_comb) {
  __shared__ __align__(16) _Float16 kbuf[2][32 * DHEAD];  // K tiles,   2x4KB
  __shared__ __align__(16) _Float16 vbuf[2][DHEAD * 32];  // V^T tiles, 2x4KB
  __shared__ __align__(16) _Float16 plds[8][16 * 32];     // per-wave P relayout, 8KB
  int tid  = threadIdx.x;
  int wave = tid >> 5, l = tid & 31;
  int itile = blockIdx.y * 8 + wave;
  if (itile > NPAD / 16 - 1) itile = NPAD / 16 - 1;       // clamp: dups store nothing
  int bh = blockIdx.x;                                    // 0..95
  int b  = bh / HEADS_, h = bh % HEADS_;
  const _Float16* qbase = qh + (size_t)bh * NPAD * DHEAD;
  const _Float16* kbase = kh + (size_t)bh * NPAD * DHEAD;
  const _Float16* vbase = vT + (size_t)bh * DHEAD * NPAD;
  int i0 = itile * 16;
  v16h aq0 = frag_a_f16(qbase, DHEAD, i0, 0);
  v16h aq1 = frag_a_f16(qbase, DHEAD, i0, 32);
  v8f acc[4] = {{}, {}, {}, {}};
  float rowm[8], rowl[8];
#pragma unroll
  for (int r = 0; r < 8; ++r) { rowm[r] = -3.0e30f; rowl[r] = 0.f; }
  int colid = l & 15;
  int hi    = l >> 4;
  _Float16* pbuf = &plds[wave][0];
  // per-thread 16B staging slots: kbuf/vbuf laid out so dst offset == tid*8 halves
  const _Float16* ksrc = kbase + (size_t)(tid >> 3) * DHEAD + (tid & 7) * 8;
  const _Float16* vsrc = vbase + (size_t)(tid >> 2) * NPAD + (tid & 3) * 8;

  auto stage = [&](int p, int j0) {
    async_cp16(&kbuf[p][tid * 8], ksrc + (size_t)j0 * DHEAD);
    async_cp16(&vbuf[p][tid * 8], vsrc + j0);
  };

  stage(0, 0);
  int pp = 0;
  for (int j0 = 0; j0 < NPAD; j0 += 32, pp ^= 1) {
    wait_async_lds();       // my async loads for buffer pp are done
    __syncthreads();        // everyone's are; also fences prior reads of pp^1
    if (j0 + 32 < NPAD) stage(pp ^ 1, j0 + 32);   // overlap next tile DMA

    const _Float16* kb = &kbuf[pp][0];
    const _Float16* vb = &vbuf[pp][0];
    // preload K fragments, then 4 score WMMAs
    v16h k00 = frag_b_f16(kb, DHEAD, 0,  0);
    v16h k01 = frag_b_f16(kb, DHEAD, 0,  32);
    v16h k10 = frag_b_f16(kb, DHEAD, 16, 0);
    v16h k11 = frag_b_f16(kb, DHEAD, 16, 32);
    v8f s0 = {}, s1 = {};
    s0 = wmma16(aq0, k00, s0);
    s1 = wmma16(aq0, k10, s1);
    s0 = wmma16(aq1, k01, s0);
    s1 = wmma16(aq1, k11, s1);

    bool ok0 = (j0 + colid) < N_;
    bool ok1 = (j0 + 16 + colid) < N_;
#pragma unroll
    for (int r = 0; r < 8; ++r) {
      if (!ok0) s0[r] = -3.0e30f;
      if (!ok1) s1[r] = -3.0e30f;
      float vmax = fmaxf(s0[r], s1[r]);
      vmax = fmaxf(vmax, __shfl_xor(vmax, 1, 16));
      vmax = fmaxf(vmax, __shfl_xor(vmax, 2, 16));
      vmax = fmaxf(vmax, __shfl_xor(vmax, 4, 16));
      vmax = fmaxf(vmax, __shfl_xor(vmax, 8, 16));
      float nm   = fmaxf(rowm[r], vmax);
      float corr = __expf(rowm[r] - nm);
      rowm[r] = nm;
      float p0 = __expf(s0[r] - nm);
      float p1 = __expf(s1[r] - nm);
      float rs = p0 + p1;
      rs += __shfl_xor(rs, 1, 16);
      rs += __shfl_xor(rs, 2, 16);
      rs += __shfl_xor(rs, 4, 16);
      rs += __shfl_xor(rs, 8, 16);
      rowl[r] = rowl[r] * corr + rs;
      s0[r] = p0;
      s1[r] = p1;
#pragma unroll
      for (int dt = 0; dt < 4; ++dt) acc[dt][r] *= corr;
    }
    // C/D-layout P -> LDS -> A-layout P (16x32 f16), wave-private region
#pragma unroll
    for (int r = 0; r < 8; ++r) {
      int mr = r + hi * 8;
      pbuf[mr * 32 + colid]      = (_Float16)s0[r];
      pbuf[mr * 32 + 16 + colid] = (_Float16)s1[r];
    }
    asm volatile("s_wait_dscnt 0x0" ::: "memory");
    v16h ap;
    {
      const _Float16* prow = pbuf + (l & 15) * 32 + hi * 8;
      v8h x0 = *(const v8h*)prow;
      v8h x1 = *(const v8h*)(prow + 16);
#pragma unroll
      for (int e = 0; e < 8; ++e) { ap[e] = x0[e]; ap[e + 8] = x1[e]; }
    }
    // P(16x32) @ V(32x64): preload V fragments, then 4 WMMAs
    v16h bv0 = frag_b_f16(vb, 32, 0,  0);
    v16h bv1 = frag_b_f16(vb, 32, 16, 0);
    v16h bv2 = frag_b_f16(vb, 32, 32, 0);
    v16h bv3 = frag_b_f16(vb, 32, 48, 0);
    acc[0] = wmma16(ap, bv0, acc[0]);
    acc[1] = wmma16(ap, bv1, acc[1]);
    acc[2] = wmma16(ap, bv2, acc[2]);
    acc[3] = wmma16(ap, bv3, acc[3]);
  }
  // epilogue: normalize and write f32 combined buffer (B*N, 768)
#pragma unroll
  for (int r = 0; r < 8; ++r) {
    int irow = i0 + r + hi * 8;
    if (irow >= N_) continue;
    float inv = 1.0f / rowl[r];
    size_t rowoff = ((size_t)b * N_ + irow) * INNER_ + (size_t)h * DHEAD;
#pragma unroll
    for (int dt = 0; dt < 4; ++dt)
      out_comb[rowoff + dt * 16 + colid] = acc[dt][r] * inv;
  }
}

// ---------------- Kernel 3: depthwise 3x3 conv, accumulate into channels 0..511
__global__ __launch_bounds__(256) void dwconv_kernel(
    const float* __restrict__ x, const float* __restrict__ w_dw,
    float* __restrict__ out_comb) {
  int idx = blockIdx.x * 256 + threadIdx.x;   // (b, ph, pw, c)
  if (idx >= B_ * HW_ * HW_ * DIM_) return;
  int c    = idx & (DIM_ - 1);
  int rest = idx >> 9;
  int pw   = rest & (HW_ - 1);
  rest >>= 5;
  int ph = rest & (HW_ - 1);
  int b  = rest >> 5;
  const float* wk = w_dw + c * 9;
  float s = 0.f;
#pragma unroll
  for (int dy = -1; dy <= 1; ++dy)
#pragma unroll
    for (int dx = -1; dx <= 1; ++dx) {
      int yy = ph + dy, xx = pw + dx;
      if (yy < 0 || yy >= HW_ || xx < 0 || xx >= HW_) continue;
      s += x[((size_t)b * N_ + 1 + yy * HW_ + xx) * DIM_ + c] *
           wk[(dy + 1) * 3 + (dx + 1)];
    }
  int n = 1 + ph * HW_ + pw;
  out_comb[((size_t)b * N_ + n) * INNER_ + c] += s;
}

// ---------------- Kernel 3b: cls row gets x[:,0,:512] added ------------------
__global__ __launch_bounds__(256) void cls_kernel(
    const float* __restrict__ x, float* __restrict__ out_comb) {
  int idx = blockIdx.x * 256 + threadIdx.x;   // B_*DIM_ = 4096
  if (idx >= B_ * DIM_) return;
  int c = idx & (DIM_ - 1);
  int b = idx >> 9;
  out_comb[(size_t)b * N_ * INNER_ + c] += x[(size_t)b * N_ * DIM_ + c];
}

// ---------------- Kernel 4: output projection + bias -------------------------
// y = out_comb(8200x768) * w_out^T(768x512) + b; wave = 32x64 block, 8 n-groups.
__global__ __launch_bounds__(256) void out_gemm_kernel(
    const float* __restrict__ out_comb, const float* __restrict__ w_out,
    const float* __restrict__ b_out, float* __restrict__ y) {
  int wave = threadIdx.x >> 5;                // n-group 0..7 (512/64)
  int m0   = blockIdx.x * 32;
  int n0   = wave * 64;
  int l    = threadIdx.x & 31;
  v8f c[2][4] = {};
  for (int k0 = 0; k0 < INNER_; k0 += 32) {
    if (k0 + 64 < INNER_) {
      int prow = m0 + (l & 15); if (prow >= M_TOT) prow = M_TOT - 1;
      __builtin_prefetch(out_comb + (size_t)prow * INNER_ + k0 + 64, 0, 1);
    }
    v16h a0 = frag_a_f32(out_comb, INNER_, m0, M_TOT, k0);
    v16h a1 = frag_a_f32(out_comb, INNER_, m0 + 16, M_TOT, k0);
    v16h b0 = frag_b_f32(w_out, INNER_, n0,      k0);
    v16h b1 = frag_b_f32(w_out, INNER_, n0 + 16, k0);
    v16h b2 = frag_b_f32(w_out, INNER_, n0 + 32, k0);
    v16h b3 = frag_b_f32(w_out, INNER_, n0 + 48, k0);
    c[0][0] = wmma16(a0, b0, c[0][0]);
    c[1][0] = wmma16(a1, b0, c[1][0]);
    c[0][1] = wmma16(a0, b1, c[0][1]);
    c[1][1] = wmma16(a1, b1, c[1][1]);
    c[0][2] = wmma16(a0, b2, c[0][2]);
    c[1][2] = wmma16(a1, b2, c[1][2]);
    c[0][3] = wmma16(a0, b3, c[0][3]);
    c[1][3] = wmma16(a1, b3, c[1][3]);
  }
  int hi = l >> 4;
#pragma unroll
  for (int i = 0; i < 2; ++i) {
#pragma unroll
    for (int j = 0; j < 4; ++j) {
      int col  = n0 + j * 16 + (l & 15);
      float bias = b_out[col];
#pragma unroll
      for (int r = 0; r < 8; ++r) {
        int m = m0 + i * 16 + r + hi * 8;
        if (m >= M_TOT) continue;
        y[(size_t)m * DIM_ + col] = c[i][j][r] + bias;
      }
    }
  }
}

extern "C" void kernel_launch(void* const* d_in, const int* in_sizes, int n_in,
                              void* d_out, int out_size, void* d_ws, size_t ws_size,
                              hipStream_t stream) {
  const float* x     = (const float*)d_in[0];
  const float* w_qkv = (const float*)d_in[1];
  const float* w_dw  = (const float*)d_in[2];
  const float* w_out = (const float*)d_in[3];
  const float* b_out = (const float*)d_in[4];
  // d_in[5], d_in[6]: h_feat = w_feat = 32 (compile-time constants here)

  size_t qelems = (size_t)B_ * HEADS_ * NPAD * DHEAD;   // f16 elements
  _Float16* qh = (_Float16*)d_ws;
  _Float16* kh = qh + qelems;
  _Float16* vT = kh + qelems;
  float* out_comb = (float*)(vT + qelems);              // (B*N, 768) f32

  // 1) QKV projection: 36 n-groups x 257 m-groups, 8 waves/block
  {
    int tiles = (QKV3 / 64) * ((M_TOT + 31) / 32);
    qkv_kernel<<<(tiles + 7) / 8, 256, 0, stream>>>(x, w_qkv, qh, kh, vT);
  }
  // 2) Flash attention: 96 (b,h) x 66 i-tiles (8 waves/block share K/V LDS tiles)
  attn_kernel<<<dim3(B_ * HEADS_, (NPAD / 16 + 7) / 8), 256, 0, stream>>>(
      qh, kh, vT, out_comb);
  // 3) depthwise conv + cls passthrough accumulate
  dwconv_kernel<<<(B_ * HW_ * HW_ * DIM_) / 256, 256, 0, stream>>>(x, w_dw, out_comb);
  cls_kernel<<<(B_ * DIM_) / 256, 256, 0, stream>>>(x, out_comb);
  // 4) output projection: 257 m-groups, 8 n-groups = 8 waves/block
  out_gemm_kernel<<<(M_TOT + 31) / 32, 256, 0, stream>>>(
      out_comb, w_out, b_out, (float*)d_out);
}
